// multi_VGAE_31430570672399
// MI455X (gfx1250) — matrix operand order repaired and checked by reading, not compiled, not run on hardware
//
#include <hip/hip_runtime.h>
#include <hip/hip_bf16.h>
#include <math.h>

// ---------------------------------------------------------------------------
// multi-VGAE forward for MI455X (gfx1250), wave32, WMMA bf16 -> f32 accum
// ---------------------------------------------------------------------------

typedef __attribute__((ext_vector_type(16))) __bf16 v16bf;
typedef __attribute__((ext_vector_type(8)))  float  v8f;

// Problem dims (fixed by the reference)
#define N1_HYB   8000
#define N2_ATAC  6000
#define RNA_CNT  5000
#define D_HYB    3000
#define D_ATAC   5000
#define H1_DIM   256
#define H2_DIM   64
#define NM_DIM   (RNA_CNT + N2_ATAC)   // 11000
#define N_ANCH   (N1_HYB - RNA_CNT)    // 3000

__device__ __forceinline__ unsigned short bfbits(float f) {
  union { float f; unsigned u; } v; v.f = f;
  unsigned r = v.u + 0x7FFFu + ((v.u >> 16) & 1u);   // round-to-nearest-even
  return (unsigned short)(r >> 16);
}

__device__ __forceinline__ __bf16 f2bf(float f) {
  union { unsigned short s; __bf16 b; } o;
  o.s = bfbits(f);
  return o.b;
}

// ---------------------------------------------------------------------------
// Staging: global fp32 -> LDS bf16.
//   As: [128][32] (k contiguous)        -> lane frags are 2x ds_load_b128
//   Bs: [BN][32] transposed (k contig)  -> lane frags are 2x ds_load_b128
// Fast path: float4 loads, packed LDS stores, no guards (interior tiles).
// Guard path: clamped addresses + select-zero (branch-free, edge tiles).
// ---------------------------------------------------------------------------
template <int BN>
__device__ __forceinline__ void stage_tile_fast(
    const float* __restrict__ A, const float* __restrict__ B,
    __bf16* __restrict__ As, __bf16* __restrict__ Bs,
    int rowBase, int colBase, int k0, int K, int N, int tid, bool pf)
{
#pragma unroll
  for (int i = 0; i < 4; ++i) {
    int idx4 = tid + i * 256;
    int m  = idx4 >> 3;            // 0..127
    int kk = (idx4 & 7) << 2;      // 0,4,..,28
    const float* ap = A + (size_t)(rowBase + m) * K + k0 + kk;
    const float4 v = *(const float4*)ap;
    ushort4 h;
    h.x = bfbits(v.x); h.y = bfbits(v.y); h.z = bfbits(v.z); h.w = bfbits(v.w);
    *(ushort4*)(As + m * 32 + kk) = h;
    if (pf) __builtin_prefetch(ap + 32, 0, 1);   // next K tile -> global_prefetch_b8
  }
  const int kp = (tid >> 4) << 1;  // even k: 0,2,..,30
#pragma unroll
  for (int q = 0; q < BN / 64; ++q) {
    int n0 = ((tid & 15) << 2) + q * 64;
    const float* bp = B + (size_t)(k0 + kp) * N + colBase + n0;
    const float4 b0 = *(const float4*)(bp);
    const float4 b1 = *(const float4*)(bp + N);
    unsigned w0 = (unsigned)bfbits(b0.x) | ((unsigned)bfbits(b1.x) << 16);
    unsigned w1 = (unsigned)bfbits(b0.y) | ((unsigned)bfbits(b1.y) << 16);
    unsigned w2 = (unsigned)bfbits(b0.z) | ((unsigned)bfbits(b1.z) << 16);
    unsigned w3 = (unsigned)bfbits(b0.w) | ((unsigned)bfbits(b1.w) << 16);
    *(unsigned*)(Bs + (n0 + 0) * 32 + kp) = w0;
    *(unsigned*)(Bs + (n0 + 1) * 32 + kp) = w1;
    *(unsigned*)(Bs + (n0 + 2) * 32 + kp) = w2;
    *(unsigned*)(Bs + (n0 + 3) * 32 + kp) = w3;
    if (pf) __builtin_prefetch(bp + (size_t)32 * N, 0, 1);
  }
}

template <int BN>
__device__ __forceinline__ void stage_tile_guard(
    const float* __restrict__ A, const float* __restrict__ B,
    __bf16* __restrict__ As, __bf16* __restrict__ Bs,
    int rowBase, int colBase, int k0, int M, int N, int K, int tid)
{
#pragma unroll
  for (int i = 0; i < 16; ++i) {
    int e  = tid + i * 256;
    int m  = e >> 5, kk = e & 31;
    int gm = rowBase + m, gk = k0 + kk;
    bool ok = (gm < M) && (gk < K);
    int cm = gm < M ? gm : M - 1;        // clamp -> always-valid address
    int ck = gk < K ? gk : K - 1;
    float v = A[(size_t)cm * K + ck];
    As[m * 32 + kk] = f2bf(ok ? v : 0.0f);
  }
#pragma unroll
  for (int i = 0; i < BN / 8; ++i) {
    int e  = tid + i * 256;
    int kk = e / BN, n = e % BN;
    int gk = k0 + kk, gn = colBase + n;
    bool ok = (gk < K) && (gn < N);
    int ck = gk < K ? gk : K - 1;
    int cn = gn < N ? gn : N - 1;
    float v = B[(size_t)ck * N + cn];
    Bs[n * 32 + kk] = f2bf(ok ? v : 0.0f);
  }
}

// Fragment build (A 16x32 / B 32x16, contiguous k = 16*lhi + i in both) + 2 x WNT WMMAs.
template <int WNT>
__device__ __forceinline__ void compute_step(
    const __bf16* __restrict__ As, const __bf16* __restrict__ Bs,
    v8f acc[][WNT], int wm, int wn, int lmod, int lhi)
{
  v16bf afrag[2], bfrag[WNT];
#pragma unroll
  for (int t = 0; t < 2; ++t) {
    const int ml = wm * 32 + t * 16 + lmod;          // A: lane = M
#pragma unroll
    for (int i = 0; i < 16; ++i) {
      int kk = lhi * 16 + i;
      afrag[t][i] = As[ml * 32 + kk];
    }
  }
#pragma unroll
  for (int t = 0; t < WNT; ++t) {
    const int nl = wn * (WNT * 16) + t * 16 + lmod;  // B: lane = N
#pragma unroll
    for (int i = 0; i < 16; ++i)
      bfrag[t][i] = Bs[nl * 32 + lhi * 16 + i];
  }
#pragma unroll
  for (int tm = 0; tm < 2; ++tm)
#pragma unroll
    for (int tn = 0; tn < WNT; ++tn)
      acc[tm][tn] = __builtin_amdgcn_wmma_f32_16x16x32_bf16(
          false, afrag[tm], false, bfrag[tn], (short)0, acc[tm][tn], false, false);
}

// ---------------------------------------------------------------------------
// Tiled GEMM: C[M,N] = act(A[M,K] @ B[K,N]); block tile 128 x (32*WNT),
// BK=32, 8 waves (4x2), per-wave 32 x (16*WNT); double-buffered LDS,
// one barrier per K step. ACT: 0=none,1=relu,2=sigmoid.
// ---------------------------------------------------------------------------
template <int ACT, int WNT>
__global__ __launch_bounds__(256)
void gemm_bf16_wmma(const float* __restrict__ A,
                    const float* __restrict__ B,
                    float* __restrict__ C,
                    int M, int N, int K)
{
  constexpr int BN = 32 * WNT;               // 128 or 256
  __shared__ __bf16 As[2][128 * 32];         // 16 KB
  __shared__ __bf16 Bs[2][BN * 32];          // 16/32 KB

  const int tid  = threadIdx.x;
  const int lane = tid & 31;
  const int wave = tid >> 5;        // 0..7
  const int wm   = wave >> 1;       // 0..3
  const int wn   = wave & 1;        // 0..1
  const int lmod = lane & 15;
  const int lhi  = lane >> 4;

  const int rowBase = blockIdx.x * 128;
  const int colBase = blockIdx.y * BN;
  const bool interior = (rowBase + 128 <= M) && (colBase + BN <= N);

  v8f acc[2][WNT] = {};
  const int nsteps = (K + 31) >> 5;

  // prologue stage into buffer 0
  if (interior && 32 <= K)
    stage_tile_fast<BN>(A, B, As[0], Bs[0], rowBase, colBase, 0, K, N, tid, nsteps > 1);
  else
    stage_tile_guard<BN>(A, B, As[0], Bs[0], rowBase, colBase, 0, M, N, K, tid);
  __syncthreads();

  for (int s = 1; s < nsteps; ++s) {
    const int k0  = s << 5;
    const int buf = s & 1;
    if (interior && (k0 + 32 <= K))
      stage_tile_fast<BN>(A, B, As[buf], Bs[buf], rowBase, colBase, k0, K, N, tid, s + 1 < nsteps);
    else
      stage_tile_guard<BN>(A, B, As[buf], Bs[buf], rowBase, colBase, k0, M, N, K, tid);
    compute_step<WNT>(As[buf ^ 1], Bs[buf ^ 1], acc, wm, wn, lmod, lhi);
    __syncthreads();
  }
  compute_step<WNT>(As[(nsteps - 1) & 1], Bs[(nsteps - 1) & 1], acc, wm, wn, lmod, lhi);

  // epilogue: D f32 16x16 layout (lane = N, VGPR r = M row)
#pragma unroll
  for (int tm = 0; tm < 2; ++tm)
#pragma unroll
    for (int tn = 0; tn < WNT; ++tn)
#pragma unroll
      for (int r = 0; r < 8; ++r) {
        int gm = rowBase + wm * 32 + tm * 16 + r + 8 * lhi;
        int gn = colBase + wn * (WNT * 16) + tn * 16 + lmod;
        if (gm < M && gn < N) {
          float v = acc[tm][tn][r];
          if (ACT == 1) v = v > 0.0f ? v : 0.0f;
          else if (ACT == 2) v = 1.0f / (1.0f + __expf(-v));
          C[(size_t)gm * N + gn] = v;
        }
      }
}

// ---------------------------------------------------------------------------
// Elementwise / reduction kernels
// ---------------------------------------------------------------------------

// Pack [W_mean | W_logstd] (each [H1,64]) into Wml [H1,128]
__global__ void pack_wml_kernel(const float* __restrict__ Wm,
                                const float* __restrict__ Wl,
                                float* __restrict__ Wml)
{
  int i = blockIdx.x * blockDim.x + threadIdx.x;
  if (i < H1_DIM * 128) {
    int r = i >> 7, c = i & 127;
    Wml[i] = (c < 64) ? Wm[r * 64 + c] : Wl[r * 64 + (c - 64)];
  }
}

// hybrid: z1 = noise*exp(logstd)+mean from fused ml [rows,128] (mean|logstd)
__global__ void sample_hyb_kernel(const float* __restrict__ noise,
                                  const float* __restrict__ ml,
                                  float* __restrict__ z1, int n)
{
  int i = blockIdx.x * blockDim.x + threadIdx.x;
  if (i < n) {
    int row = i >> 6, col = i & 63;
    float mean = ml[row * 128 + col];
    float ls   = ml[row * 128 + 64 + col];
    z1[i] = noise[i] * expf(ls) + mean;
  }
}

// atac: split fused ml into mean/logstd outputs and compute z2
__global__ void sample_atac_kernel(const float* __restrict__ noise,
                                   const float* __restrict__ ml,
                                   float* __restrict__ meanA,
                                   float* __restrict__ logsA,
                                   float* __restrict__ z2, int n)
{
  int i = blockIdx.x * blockDim.x + threadIdx.x;
  if (i < n) {
    int row = i >> 6, col = i & 63;
    float mean = ml[row * 128 + col];
    float ls   = ml[row * 128 + 64 + col];
    meanA[i] = mean;
    logsA[i] = ls;
    z2[i] = noise[i] * expf(ls) + mean;
  }
}

__global__ void zm_concat_kernel(const float* __restrict__ z1,
                                 const float* __restrict__ z2,
                                 float* __restrict__ zm)
{
  int i = blockIdx.x * blockDim.x + threadIdx.x;
  int n = NM_DIM * H2_DIM;
  if (i < n) {
    int row = i / H2_DIM;
    zm[i] = (row < RNA_CNT) ? z1[i] : z2[i - RNA_CNT * H2_DIM];
  }
}

__global__ void transpose_kernel(const float* __restrict__ zm,
                                 float* __restrict__ zt)
{
  int i = blockIdx.x * blockDim.x + threadIdx.x;   // over NM*H2
  int n = NM_DIM * H2_DIM;
  if (i < n) {
    int row = i >> 6;          // H2_DIM == 64
    int k   = i & 63;
    zt[(size_t)k * NM_DIM + row] = zm[i];
  }
}

__global__ void log_softmax_kernel(const float* __restrict__ zm,
                                   float* __restrict__ out)
{
  int r = blockIdx.x * blockDim.x + threadIdx.x;
  if (r >= NM_DIM) return;
  const float* x = zm + (size_t)r * H2_DIM;
  float mx = -1e30f;
#pragma unroll
  for (int j = 0; j < H2_DIM; ++j) mx = fmaxf(mx, x[j]);
  float s = 0.0f;
#pragma unroll
  for (int j = 0; j < H2_DIM; ++j) s += expf(x[j] - mx);
  float lse = mx + logf(s);
  float* o = out + (size_t)r * H2_DIM;
#pragma unroll
  for (int j = 0; j < H2_DIM; ++j) o[j] = x[j] - lse;
}

__global__ void anchor_norm_kernel(const float* __restrict__ z1,
                                   const float* __restrict__ z2,
                                   const int* __restrict__ anchor,
                                   float* __restrict__ norms)
{
  int r = blockIdx.x * blockDim.x + threadIdx.x;
  if (r >= N_ANCH) return;
  const float* a = z1 + (size_t)(RNA_CNT + r) * H2_DIM;
  const float* b = z2 + (size_t)(anchor[r] - RNA_CNT) * H2_DIM;
  float s = 0.0f;
#pragma unroll
  for (int j = 0; j < H2_DIM; ++j) {
    float d = a[j] - b[j];
    s += d * d;
  }
  norms[r] = sqrtf(s);
}

// Deterministic single-block reduction (fixed summation order, no atomics).
__global__ void loss_reduce_kernel(const float* __restrict__ norms,
                                   float* __restrict__ loss_out)
{
  __shared__ float sh[256];
  int t = threadIdx.x;
  float s = 0.0f;
  for (int i = t; i < N_ANCH; i += 256) s += norms[i];
  sh[t] = s;
  __syncthreads();
  for (int stride = 128; stride > 0; stride >>= 1) {
    if (t < stride) sh[t] += sh[t + stride];
    __syncthreads();
  }
  if (t == 0) loss_out[0] = sh[0] / (float)N_ANCH;
}

// ---------------------------------------------------------------------------
// Launcher
// ---------------------------------------------------------------------------
extern "C" void kernel_launch(void* const* d_in, const int* in_sizes, int n_in,
                              void* d_out, int out_size, void* d_ws, size_t ws_size,
                              hipStream_t stream)
{
  const float* X1   = (const float*)d_in[0];
  const float* X2   = (const float*)d_in[1];
  const float* adj1 = (const float*)d_in[2];
  const float* adj2 = (const float*)d_in[3];
  const float* Wb1  = (const float*)d_in[4];
  const float* Wm1  = (const float*)d_in[5];
  const float* Wl1  = (const float*)d_in[6];
  const float* Wb2  = (const float*)d_in[7];
  const float* Wm2  = (const float*)d_in[8];
  const float* Wl2  = (const float*)d_in[9];
  const float* noise1 = (const float*)d_in[10];
  const float* noise2 = (const float*)d_in[11];
  const int*   anchor = (const int*)d_in[12];
  (void)in_sizes; (void)n_in; (void)ws_size;

  // ---- output layout (flat fp32, reference return order) ----
  float* out = (float*)d_out;
  const size_t szA    = (size_t)NM_DIM * NM_DIM;      // A_pred_m
  const size_t szZm   = (size_t)NM_DIM * H2_DIM;      // Z_m
  const size_t szZ1   = (size_t)N1_HYB * H2_DIM;      // sampled_z_1
  const size_t szMa   = (size_t)N2_ATAC * H2_DIM;     // mean_atac
  float* out_Apred  = out;
  float* out_Zm     = out_Apred + szA;
  float* out_z1     = out_Zm + szZm;
  float* out_meanA  = out_z1 + szZ1;
  float* out_logsA  = out_meanA + szMa;
  float* out_lsm    = out_logsA + szMa;
  float* out_loss   = out_lsm + szZm;
  (void)out_size;

  // ---- workspace layout (floats) ----
  float* ws = (float*)d_ws;
  float* tmp1    = ws;                                   // [8000,256]
  float* hidden1 = tmp1    + (size_t)N1_HYB * H1_DIM;    // [8000,256]
  float* hml1    = hidden1 + (size_t)N1_HYB * H1_DIM;    // [8000,128]
  float* ml1     = hml1    + (size_t)N1_HYB * 128;       // [8000,128]
  float* tmp2    = ml1     + (size_t)N1_HYB * 128;       // [6000,256]
  float* hidden2 = tmp2    + (size_t)N2_ATAC * H1_DIM;   // [6000,256]
  float* hml2    = hidden2 + (size_t)N2_ATAC * H1_DIM;   // [6000,128]
  float* ml2     = hml2    + (size_t)N2_ATAC * 128;      // [6000,128]
  float* z2      = ml2     + (size_t)N2_ATAC * 128;      // [6000,64]
  float* Zt      = z2      + (size_t)N2_ATAC * H2_DIM;   // [64,11000]
  float* Wml1    = Zt      + (size_t)H2_DIM * NM_DIM;    // [256,128]
  float* Wml2    = Wml1    + (size_t)H1_DIM * 128;       // [256,128]
  float* norms   = Wml2    + (size_t)H1_DIM * 128;       // [3000]

  const dim3 blk(256);
  auto gx = [](int m) { return (m + 127) / 128; };

  // ---- pack fused [W_mean|W_logstd] weights ----
  pack_wml_kernel<<<(H1_DIM * 128 + 255) / 256, blk, 0, stream>>>(Wm1, Wl1, Wml1);
  pack_wml_kernel<<<(H1_DIM * 128 + 255) / 256, blk, 0, stream>>>(Wm2, Wl2, Wml2);

  // ---- hybrid branch ----
  gemm_bf16_wmma<0, 8><<<dim3(gx(N1_HYB), 1), blk, 0, stream>>>(X1,   Wb1,  tmp1,    N1_HYB, H1_DIM, D_HYB);
  gemm_bf16_wmma<1, 8><<<dim3(gx(N1_HYB), 1), blk, 0, stream>>>(adj1, tmp1, hidden1, N1_HYB, H1_DIM, N1_HYB);
  gemm_bf16_wmma<0, 4><<<dim3(gx(N1_HYB), 1), blk, 0, stream>>>(hidden1, Wml1, hml1, N1_HYB, 128, H1_DIM);
  gemm_bf16_wmma<0, 4><<<dim3(gx(N1_HYB), 1), blk, 0, stream>>>(adj1, hml1, ml1,     N1_HYB, 128, N1_HYB);

  // ---- atac branch ----
  gemm_bf16_wmma<0, 8><<<dim3(gx(N2_ATAC), 1), blk, 0, stream>>>(X2,   Wb2,  tmp2,    N2_ATAC, H1_DIM, D_ATAC);
  gemm_bf16_wmma<1, 8><<<dim3(gx(N2_ATAC), 1), blk, 0, stream>>>(adj2, tmp2, hidden2, N2_ATAC, H1_DIM, N2_ATAC);
  gemm_bf16_wmma<0, 4><<<dim3(gx(N2_ATAC), 1), blk, 0, stream>>>(hidden2, Wml2, hml2, N2_ATAC, 128, H1_DIM);
  gemm_bf16_wmma<0, 4><<<dim3(gx(N2_ATAC), 1), blk, 0, stream>>>(adj2, hml2, ml2,     N2_ATAC, 128, N2_ATAC);

  // ---- reparameterize (+ split atac mean/logstd into outputs) ----
  {
    int n1 = N1_HYB * H2_DIM;
    sample_hyb_kernel<<<(n1 + 255) / 256, blk, 0, stream>>>(noise1, ml1, out_z1, n1);
    int n2 = N2_ATAC * H2_DIM;
    sample_atac_kernel<<<(n2 + 255) / 256, blk, 0, stream>>>(noise2, ml2, out_meanA, out_logsA, z2, n2);
  }

  // ---- Z_m = concat(z1[:RNA], z2); Zt = Z_m^T ----
  {
    int n = NM_DIM * H2_DIM;
    zm_concat_kernel<<<(n + 255) / 256, blk, 0, stream>>>(out_z1, z2, out_Zm);
    transpose_kernel<<<(n + 255) / 256, blk, 0, stream>>>(out_Zm, Zt);
  }

  // ---- decoder: A_pred = sigmoid(Z_m @ Z_m^T), 128x128 tiles ----
  gemm_bf16_wmma<2, 4><<<dim3(gx(NM_DIM), (NM_DIM + 127) / 128), blk, 0, stream>>>(
      out_Zm, Zt, out_Apred, NM_DIM, NM_DIM, H2_DIM);

  // ---- log_softmax(Z_m) ----
  log_softmax_kernel<<<(NM_DIM + 255) / 256, blk, 0, stream>>>(out_Zm, out_lsm);

  // ---- anchor alignment loss (deterministic two-stage reduce) ----
  anchor_norm_kernel<<<(N_ANCH + 255) / 256, blk, 0, stream>>>(out_z1, z2, anchor, norms);
  loss_reduce_kernel<<<1, blk, 0, stream>>>(norms, out_loss);
}
